// TGCNModel_65592740544791
// MI455X (gfx1250) — matrix operand order repaired and checked by reading, not compiled, not run on hardware
//
#include <hip/hip_runtime.h>
#include <hip/hip_bf16.h>

typedef __attribute__((ext_vector_type(16))) _Float16 v16h;
typedef __attribute__((ext_vector_type(8)))  _Float16 v8h;
typedef __attribute__((ext_vector_type(8)))  float    v8f;
typedef __attribute__((ext_vector_type(4)))  float    v4f;

#define B_    8
#define T_    12
#define N_    4000
#define C_    2
#define H_    64
#define BT_   (B_ * T_)        // 96
#define JCOLS (BT_ * C_)       // 192  (columns of the aggregation GEMM)
#define ROWS  (B_ * N_)        // 32000 (LSTM batch)
#define G4    (4 * H_)         // 256  (gate width)
#define KCAT  (2 * H_)         // 128  (concatenated LSTM K)

// GEMM1 tiling: 4 waves/block, 3 n-tiles per wave (4*3*16 = 192 = JCOLS)
#define G1_WAVES 4
#define G1_NT    3

// ---------------------------------------------------------------------------
// Zero the recurrent states (ws is poisoned; must re-init every call).
__global__ void k_init(float* __restrict__ cst, _Float16* __restrict__ h16, int n) {
    int i = blockIdx.x * blockDim.x + threadIdx.x;
    if (i < n) { cst[i] = 0.0f; h16[i] = (_Float16)0.0f; }
}

// ---------------------------------------------------------------------------
// adj f32 -> f16 (read 64MB once, write 32MB once; saves 12x re-read in f32).
__global__ void k_cvt_adj(const float* __restrict__ adj, _Float16* __restrict__ adjh,
                          int n) {
    int i = blockIdx.x * blockDim.x + threadIdx.x;
    if (i < n) adjh[i] = (_Float16)adj[i];
}

// ---------------------------------------------------------------------------
// Pack x[B,T,N,C] (f32) into column-major f16 xpT[j][k], j=(b*T+t)*C+c, k=node.
__global__ void k_pack(const float* __restrict__ x, _Float16* __restrict__ xpT) {
    int i = blockIdx.x * blockDim.x + threadIdx.x;   // i = j*N_ + k
    if (i >= JCOLS * N_) return;
    int j = i / N_;
    int k = i - j * N_;
    int bt = j >> 1;
    int c  = j & 1;
    xpT[i] = (_Float16)x[((size_t)bt * N_ + k) * C_ + c];
}

// ---------------------------------------------------------------------------
// Pre-pack LSTM weights: Wcat16[col][k], k<64 -> W_ih[col][k], else W_hh[col][k-64];
// bsum[col] = b_ih[col] + b_hh[col].
__global__ void k_prepw(const float* __restrict__ Wih, const float* __restrict__ Whh,
                        const float* __restrict__ bih, const float* __restrict__ bhh,
                        _Float16* __restrict__ Wcat16, float* __restrict__ bsum) {
    int i = blockIdx.x * blockDim.x + threadIdx.x;   // col*KCAT + k
    if (i >= G4 * KCAT) return;
    int col = i / KCAT;
    int k   = i - col * KCAT;
    float v = (k < H_) ? Wih[(size_t)col * H_ + k] : Whh[(size_t)col * H_ + (k - H_)];
    Wcat16[i] = (_Float16)v;
    if (k == 0) bsum[col] = bih[col] + bhh[col];
}

// ---------------------------------------------------------------------------
// xg[n, j] = sum_m adjh[n,m] * xpT[j][m]      (M=4000, N=192, K=4000, f16 WMMA)
// Block = 128 threads (4 waves); each wave owns 3 n-tiles, so each A-fragment
// (the bandwidth-critical adj stream) feeds 3 WMMAs.
__global__ void __launch_bounds__(G1_WAVES * 32)
k_gemm_adj(const _Float16* __restrict__ adjh, const _Float16* __restrict__ xpT,
           float* __restrict__ xg) {
    const int m0     = blockIdx.x * 16;
    const int wave   = threadIdx.x >> 5;
    const int lane   = threadIdx.x & 31;
    const int half16 = lane >> 4;
    const int l15    = lane & 15;
    const int aoff   = half16 * 16;       // A: K 0-15 / 16-31
    const int boff   = half16 * 16;       // B: K 0-15 / 16-31

    const _Float16* arow = adjh + (size_t)(m0 + l15) * N_;   // A row, f16
    const _Float16* bcol[G1_NT];
#pragma unroll
    for (int j = 0; j < G1_NT; ++j) {
        int n0 = (wave * G1_NT + j) * 16;
        bcol[j] = xpT + (size_t)(n0 + l15) * N_;
    }

    v8f acc[G1_NT];
#pragma unroll
    for (int j = 0; j < G1_NT; ++j) acc[j] = (v8f){};

    for (int k0 = 0; k0 < N_; k0 += 32) {
        // prefetch next adj chunk of this row into the cache hierarchy
        if (k0 + 256 < N_) __builtin_prefetch(arow + k0 + 256, 0, 1);
        v16h a;
        {
            v8h lo = *(const v8h*)(arow + k0 + aoff);
            v8h hi = *(const v8h*)(arow + k0 + 8 + aoff);
#pragma unroll
            for (int i = 0; i < 8; ++i) { a[i] = lo[i]; a[8 + i] = hi[i]; }
        }
#pragma unroll
        for (int j = 0; j < G1_NT; ++j) {
            v16h b = *(const v16h*)(bcol[j] + k0 + boff);
            acc[j] = __builtin_amdgcn_wmma_f32_16x16x32_f16(
                false, a, false, b, (short)0, acc[j], false, false);
        }
    }
    // D layout: VGPR v -> row m0+v (+8 for upper lanes), col n0 + (lane&15)
#pragma unroll
    for (int j = 0; j < G1_NT; ++j) {
        int col = (wave * G1_NT + j) * 16 + l15;
#pragma unroll
        for (int v = 0; v < 8; ++v) {
            int row = m0 + v + half16 * 8;
            xg[(size_t)row * JCOLS + col] = acc[j][v];
        }
    }
}

// ---------------------------------------------------------------------------
// g = relu(xg @ W_gcn^T + b_gcn), stored as f16 sequence gseq[t][b*N+n][h].
__global__ void k_gcn(const float* __restrict__ xg, const float* __restrict__ Wg,
                      const float* __restrict__ bg, _Float16* __restrict__ gseq) {
    long long i = (long long)blockIdx.x * blockDim.x + threadIdx.x; // n*BT*H+bt*H+h
    if (i >= (long long)N_ * BT_ * H_) return;
    int h  = (int)(i % H_);
    long long r = i / H_;
    int bt = (int)(r % BT_);
    int n  = (int)(r / BT_);
    int b  = bt / T_;
    int t  = bt % T_;
    float v0 = xg[(size_t)n * JCOLS + bt * 2 + 0];
    float v1 = xg[(size_t)n * JCOLS + bt * 2 + 1];
    float val = fmaf(v0, Wg[h * 2 + 0], fmaf(v1, Wg[h * 2 + 1], bg[h]));
    val = val > 0.0f ? val : 0.0f;
    gseq[((size_t)t * ROWS + (size_t)b * N_ + n) * H_ + h] = (_Float16)val;
}

// ---------------------------------------------------------------------------
// One LSTM timestep. Block = 16 batch rows x 256 gate cols (8 waves x 32 cols).
// gates = [g_t | h16] (K=128, f16) x Wcat16 via 8 WMMAs per wave, staged to
// LDS, then the fused sigmoid/tanh cell update.
__global__ void __launch_bounds__(256)
k_lstm(const _Float16* __restrict__ gt, _Float16* __restrict__ h16,
       float* __restrict__ cst,
       const _Float16* __restrict__ Wcat16, const float* __restrict__ bsum) {
    __shared__ float lds[16 * G4];   // 16 KB

    const int r0     = blockIdx.x * 16;
    const int wave   = threadIdx.x >> 5;
    const int lane   = threadIdx.x & 31;
    const int half16 = lane >> 4;
    const int l15    = lane & 15;
    const int aoff   = half16 * 16;
    const int boff   = half16 * 16;

    const _Float16* grow = gt  + (size_t)(r0 + l15) * H_;
    const _Float16* hrow = h16 + (size_t)(r0 + l15) * H_;

    v8f acc0 = {}, acc1 = {};
#pragma unroll
    for (int ki = 0; ki < 4; ++ki) {                 // K = 128 in 4 chunks of 32
        const _Float16* src = (ki < 2) ? grow : hrow;
        const int ks = (ki & 1) * 32;                // offset inside the source
        const int kg = ki * 32;                      // offset in concatenated K
        v16h a;
        {
            v8h lo = *(const v8h*)(src + ks + aoff);
            v8h hi = *(const v8h*)(src + ks + 8 + aoff);
#pragma unroll
            for (int i = 0; i < 8; ++i) { a[i] = lo[i]; a[8 + i] = hi[i]; }
        }
#pragma unroll
        for (int nt = 0; nt < 2; ++nt) {
            const int col = wave * 32 + nt * 16 + l15;       // gate column
            v16h bm = *(const v16h*)(Wcat16 + (size_t)col * KCAT + kg + boff);
            if (nt == 0)
                acc0 = __builtin_amdgcn_wmma_f32_16x16x32_f16(
                    false, a, false, bm, (short)0, acc0, false, false);
            else
                acc1 = __builtin_amdgcn_wmma_f32_16x16x32_f16(
                    false, a, false, bm, (short)0, acc1, false, false);
        }
    }

    // bias + stage gates to LDS (each D VGPR shares one column per lane)
#pragma unroll
    for (int nt = 0; nt < 2; ++nt) {
        v8f acc = nt ? acc1 : acc0;
        int col = wave * 32 + nt * 16 + l15;
        float bias = bsum[col];
#pragma unroll
        for (int v = 0; v < 8; ++v)
            lds[(v + half16 * 8) * G4 + col] = acc[v] + bias;
    }
    __syncthreads();

    // cell update: 16 rows x 64 units = 1024 items over 256 threads
    for (int idx = threadIdx.x; idx < 16 * H_; idx += 256) {
        int rl = idx >> 6;
        int u  = idx & 63;
        float ig = lds[rl * G4 + u];
        float fg = lds[rl * G4 + 64 + u];
        float gg = lds[rl * G4 + 128 + u];
        float og = lds[rl * G4 + 192 + u];
        size_t gi = (size_t)(r0 + rl) * H_ + u;
        float c  = cst[gi];
        float si = 1.0f / (1.0f + __expf(-ig));
        float sf = 1.0f / (1.0f + __expf(-fg));
        float so = 1.0f / (1.0f + __expf(-og));
        c = sf * c + si * tanhf(gg);
        float hn = so * tanhf(c);
        cst[gi] = c;
        h16[gi] = (_Float16)hn;
    }
}

// ---------------------------------------------------------------------------
// out[b*N+n] = dot(h_last, W_out) + b_out   (O = 1)
__global__ void k_out(const _Float16* __restrict__ h16, const float* __restrict__ Wout,
                      const float* __restrict__ bout, float* __restrict__ out) {
    int r = blockIdx.x * blockDim.x + threadIdx.x;
    if (r >= ROWS) return;
    const _Float16* hr = h16 + (size_t)r * H_;
    float s = bout[0];
#pragma unroll
    for (int h = 0; h < H_; ++h) s = fmaf((float)hr[h], Wout[h], s);
    out[r] = s;
}

// ---------------------------------------------------------------------------
extern "C" void kernel_launch(void* const* d_in, const int* in_sizes, int n_in,
                              void* d_out, int out_size, void* d_ws, size_t ws_size,
                              hipStream_t stream) {
    const float* x    = (const float*)d_in[0];
    const float* adj  = (const float*)d_in[1];
    const float* Wg   = (const float*)d_in[2];
    const float* bg   = (const float*)d_in[3];
    const float* Wih  = (const float*)d_in[4];
    const float* Whh  = (const float*)d_in[5];
    const float* bih  = (const float*)d_in[6];
    const float* bhh  = (const float*)d_in[7];
    const float* Wout = (const float*)d_in[8];
    const float* bout = (const float*)d_in[9];
    float* out = (float*)d_out;
    (void)in_sizes; (void)n_in; (void)out_size; (void)ws_size;

    char* w = (char*)d_ws;
    size_t off = 0;
    auto take = [&](size_t bytes) -> char* {
        char* p = w + off;
        off = (off + bytes + 255) & ~(size_t)255;
        return p;
    };
    _Float16* adjh  = (_Float16*)take((size_t)N_ * N_ * sizeof(_Float16));     // 32 MB
    _Float16* xpT   = (_Float16*)take((size_t)JCOLS * N_ * sizeof(_Float16));  // 1.5 MB
    float*    xg    = (float*)   take((size_t)N_ * JCOLS * sizeof(float));     // 3 MB
    _Float16* gseq  = (_Float16*)take((size_t)T_ * ROWS * H_ * sizeof(_Float16)); // 49 MB
    _Float16* h16   = (_Float16*)take((size_t)ROWS * H_ * sizeof(_Float16));   // 4 MB
    float*    cst   = (float*)   take((size_t)ROWS * H_ * sizeof(float));      // 8 MB
    _Float16* Wcat16= (_Float16*)take((size_t)G4 * KCAT * sizeof(_Float16));   // 64 KB
    float*    bsum  = (float*)   take((size_t)G4 * sizeof(float));

    {   // zero recurrent state
        int n = ROWS * H_;
        k_init<<<(n + 255) / 256, 256, 0, stream>>>(cst, h16, n);
    }
    {   // adj -> f16 (single pass)
        int n = N_ * N_;
        k_cvt_adj<<<(n + 255) / 256, 256, 0, stream>>>(adj, adjh, n);
    }
    {   // pack x -> column-major f16
        int n = JCOLS * N_;
        k_pack<<<(n + 255) / 256, 256, 0, stream>>>(x, xpT);
    }
    {   // pre-pack LSTM weights + bias sums
        int n = G4 * KCAT;
        k_prepw<<<(n + 255) / 256, 256, 0, stream>>>(Wih, Whh, bih, bhh, Wcat16, bsum);
    }
    // graph aggregation GEMM: 250 blocks x 4 waves x 3 n-tiles (adj streamed once)
    k_gemm_adj<<<N_ / 16, G1_WAVES * 32, 0, stream>>>(adjh, xpT, xg);
    {   // GCN linear + relu -> f16 sequence
        long long n = (long long)N_ * BT_ * H_;
        k_gcn<<<(unsigned)((n + 255) / 256), 256, 0, stream>>>(xg, Wg, bg, gseq);
    }
    // recurrent LSTM: one launch per timestep (stream order = recurrence order)
    for (int t = 0; t < T_; ++t) {
        const _Float16* gt = gseq + (size_t)t * ROWS * H_;
        k_lstm<<<ROWS / 16, 256, 0, stream>>>(gt, h16, cst, Wcat16, bsum);
    }
    // output projection
    k_out<<<(ROWS + 255) / 256, 256, 0, stream>>>(h16, Wout, bout, out);
}
